// rpn_29824252903846
// MI455X (gfx1250) — matrix operand order repaired and checked by reading, not compiled, not run on hardware
//
#include <hip/hip_runtime.h>
#include <stdint.h>
#include <math.h>

// ---------------- problem constants (from setup_inputs) ----------------
#define NIMG   8
#define CIN    512
#define CMID   512
#define FH     64
#define FW     64
#define KDIM   (CIN * 9)        // 4608
#define NPIX   (FH * FW)        // 4096
#define NANCH  (NPIX * 9)       // 36864
#define PRE_NMS   1000
#define POST_NMS  300
#define NMS_THR   0.7f
#define NEG_INF_F (-1000000000.0f)

#define LDSTRIDE 36             // ushort row stride in LDS: 72B -> 16 rows hit 16 banks

// ---------------- WMMA / TDM types ----------------
typedef __attribute__((ext_vector_type(16))) __bf16 v16bf;
typedef __attribute__((ext_vector_type(8)))  float  v8f;
typedef __attribute__((ext_vector_type(4)))  unsigned int u32x4;
typedef __attribute__((ext_vector_type(8)))  int i32x8;
typedef __attribute__((ext_vector_type(4)))  int i32x4;
typedef unsigned short ushort_t;

// ---------------- bf16 split helpers ----------------
__device__ __forceinline__ unsigned short f2bf_bits(float f) {
    unsigned u = __float_as_uint(f);
    unsigned r = u + 0x7FFFu + ((u >> 16) & 1u);   // round-to-nearest-even
    return (unsigned short)(r >> 16);
}
__device__ __forceinline__ float bf2f_bits(unsigned short h) {
    return __uint_as_float(((unsigned)h) << 16);
}

// ---------------- Tensor Data Mover: 2D tile -> LDS (bf16 elements) ----------
// D# per CDNA5 ISA 8.x: group0 {count,lds_addr,global_addr,type=2},
// group1 {data_size=2B, pad_enable, pad_interval, pad_amount, dims/strides}.
// pad_interval_code=3 -> pad after 16 DWORDs (64B row), pad_amount_code=1 -> +2 DWORDs (8B)
// giving the 72B (36-ushort) LDS row stride the fragment loaders expect.
// NOTE: this toolchain exposes the 6-arg builtin:
//   (u32x4 g0, i32x8 g1, i32x4 g2, i32x4 g3, i32x8 extra, i32 cpol)
__device__ __forceinline__ void tdm_load_2d_bf16(
    const void* gsrc, unsigned lds_byte_off,
    unsigned tile_k, unsigned tile_rows,
    unsigned tensor_k, unsigned tensor_rows, unsigned row_stride_elems)
{
    unsigned long long ga = (unsigned long long)gsrc;
    u32x4 g0;
    g0[0] = 1u;                                            // count=1 (valid user D#)
    g0[1] = lds_byte_off;                                  // lds_addr
    g0[2] = (unsigned)(ga & 0xFFFFFFFFull);                // global_addr[31:0]
    g0[3] = (unsigned)((ga >> 32) & 0x01FFFFFFull) | (2u << 30); // addr[56:32] | type=2
    i32x8 g1;
    g1[0] = (int)((1u << 16) | (1u << 20) | (3u << 22) | (1u << 25));
    g1[1] = (int)((tensor_k & 0xFFFFu) << 16);             // tensor_dim0[15:0]
    g1[2] = (int)((tensor_k >> 16) | ((tensor_rows & 0xFFFFu) << 16));
    g1[3] = (int)((tensor_rows >> 16) | (tile_k << 16));   // tile_dim0
    g1[4] = (int)(tile_rows & 0xFFFFu);                    // tile_dim1 (tile_dim2=0)
    g1[5] = (int)row_stride_elems;                         // tensor_dim0_stride
    g1[6] = 0;
    g1[7] = 0;
    i32x4 gz4; gz4[0] = 0; gz4[1] = 0; gz4[2] = 0; gz4[3] = 0;   // 2D: groups 2/3 unused
    i32x8 gz8;
#pragma unroll
    for (int i = 0; i < 8; ++i) gz8[i] = 0;
    __builtin_amdgcn_tensor_load_to_lds(g0, g1, gz4, gz4, gz8, 0);
}
__device__ __forceinline__ unsigned lds_off(const void* p) {
    return (unsigned)(unsigned long long)p;                // low 32 bits = LDS byte offset
}

// ---------------- WMMA fragment loaders (CDNA5 16-bit layouts) ----------------
// A 16x32 (16-bit): lanes 0-15 -> M=lane, K in {0..7,16..23}; lanes 16-31 -> K in {8..15,24..31}
__device__ __forceinline__ v16bf load_frag_a(const unsigned short* lds, int row, int lane) {
    union { v16bf v; unsigned u[8]; } f;
    const unsigned short* p = lds + row * LDSTRIDE;
    const int kb = (lane & 16) ? 8 : 0;
#pragma unroll
    for (int v = 0; v < 4; ++v) f.u[v]     = *(const unsigned*)(p + kb + 2 * v);
#pragma unroll
    for (int v = 0; v < 4; ++v) f.u[4 + v] = *(const unsigned*)(p + kb + 16 + 2 * v);
    return f.v;
}
// B 32x16 (16-bit): lanes 0-15 -> N=lane, K in {0..7,16..23}; lanes 16-31 -> K in {8..15,24..31} (LDS stored [n][k])
__device__ __forceinline__ v16bf load_frag_b(const unsigned short* lds, int row, int lane) {
    union { v16bf v; unsigned u[8]; } f;
    const unsigned short* p = lds + row * LDSTRIDE + ((lane >> 4) * 8);
#pragma unroll
    for (int v = 0; v < 8; ++v) f.u[v] = *(const unsigned*)(p + 2 * (v & 3) + 16 * (v >> 2));
    return f.v;
}

#define WMMA_BF16(A, B, C) \
    __builtin_amdgcn_wmma_f32_16x16x32_bf16(false, (A), false, (B), (short)0, (C), false, false)

// =====================================================================
// Kernel 0a: one-shot f32 -> (bf16 hi, bf16 lo) split (bandwidth bound)
// =====================================================================
__global__ __launch_bounds__(256) void split_pair(
    const float* __restrict__ src, ushort_t* __restrict__ hi,
    ushort_t* __restrict__ lo, int count)
{
    int i = blockIdx.x * 256 + threadIdx.x;
    if (i >= count) return;
    float v = src[i];
    unsigned short h = f2bf_bits(v);
    hi[i] = h;
    lo[i] = f2bf_bits(v - bf2f_bits(h));
}

// Kernel 0b: pack cls(18)+reg(36) weights into padded [64][512] bf16 hi/lo
__global__ __launch_bounds__(256) void build_head_w(
    const float* __restrict__ Wc, const float* __restrict__ Wr,
    ushort_t* __restrict__ WoHi, ushort_t* __restrict__ WoLo)
{
    int i = blockIdx.x * 256 + threadIdx.x;
    if (i >= 64 * CMID) return;
    int o = i >> 9, k = i & 511;
    float v = 0.f;
    if (o < 18)      v = Wc[(size_t)o * CMID + k];
    else if (o < 54) v = Wr[(size_t)(o - 18) * CMID + k];
    unsigned short h = f2bf_bits(v);
    WoHi[i] = h;
    WoLo[i] = f2bf_bits(v - bf2f_bits(h));
}

// =====================================================================
// Kernel 1: 3x3 conv (SAME) + bias + ReLU, implicit GEMM, bf16 hi/lo split.
//   A tile: manual zero-padded im2col gather from pre-split x planes.
//   B tile: TDM tensor_load_to_lds (hi+lo descriptors) issued by wave 0.
//   Output written directly as bf16 hi/lo planes for the head GEMM.
//   grid: (N/128=4, 64 h-rows, batch=8), block 256 (8 waves 2x4)
// =====================================================================
__global__ __launch_bounds__(256) void conv3x3_wmma(
    const ushort_t* __restrict__ xHi, const ushort_t* __restrict__ xLo,
    const ushort_t* __restrict__ WbHi, const ushort_t* __restrict__ WbLo,
    const float* __restrict__ bb,
    ushort_t* __restrict__ baseHi, ushort_t* __restrict__ baseLo)
{
    __shared__ __align__(16) unsigned short sAH[64 * LDSTRIDE];
    __shared__ __align__(16) unsigned short sAL[64 * LDSTRIDE];
    __shared__ __align__(16) unsigned short sBH[128 * LDSTRIDE];
    __shared__ __align__(16) unsigned short sBL[128 * LDSTRIDE];

    const int tid   = threadIdx.x;
    const int batch = blockIdx.z;
    const int h     = blockIdx.y;          // m-tile == one image row
    const int n0    = blockIdx.x * 128;    // output-channel base

    const int wave = tid >> 5, lane = tid & 31;
    const int wm = wave >> 2;              // 0..1 -> m offset 32*wm
    const int wn = wave & 3;               // 0..3 -> n offset 32*wn

    const unsigned sBH_off = lds_off(sBH);
    const unsigned sBL_off = lds_off(sBL);

    const v8f vzero = {0.f, 0.f, 0.f, 0.f, 0.f, 0.f, 0.f, 0.f};
    v8f acc[2][2];
#pragma unroll
    for (int mt = 0; mt < 2; ++mt)
#pragma unroll
        for (int nt = 0; nt < 2; ++nt) acc[mt][nt] = vzero;

    for (int kb = 0; kb < KDIM / 32; ++kb) {
        __syncthreads();                   // LDS free for refill
        if (wave == 0) {                   // async DMA: weight tile 128 x 32 (hi+lo)
            const size_t goff = (size_t)n0 * KDIM + kb * 32;
            tdm_load_2d_bf16(WbHi + goff, sBH_off, 32, 128, KDIM, CMID, KDIM);
            tdm_load_2d_bf16(WbLo + goff, sBL_off, 32, 128, KDIM, CMID, KDIM);
        }
        // ---- A tile 64(m=w) x 32(k): padded patch gather from bf16 planes
#pragma unroll
        for (int it = 0; it < 8; ++it) {
            int idx  = it * 256 + tid;
            int wpix = idx & 63;
            int kk   = idx >> 6;
            int k    = kb * 32 + kk;
            int ci   = k / 9;
            int r    = k - ci * 9;
            int dy   = r / 3 - 1, dx = r - (r / 3) * 3 - 1;
            int hh = h + dy, ww = wpix + dx;
            unsigned short hi = 0, lo = 0;
            if (hh >= 0 && hh < FH && ww >= 0 && ww < FW) {
                size_t off = (((size_t)batch * CIN + ci) * FH + hh) * FW + ww;
                hi = xHi[off];
                lo = xLo[off];
            }
            sAH[wpix * LDSTRIDE + kk] = hi;
            sAL[wpix * LDSTRIDE + kk] = lo;
        }
        if (wave == 0) __builtin_amdgcn_s_wait_tensorcnt(0);
        __syncthreads();                   // publish TDM + A fill to all waves

        v16bf aH[2], aL[2], bH[2], bL[2];
#pragma unroll
        for (int mt = 0; mt < 2; ++mt) {
            int row = wm * 32 + mt * 16 + (lane & 15);
            aH[mt] = load_frag_a(sAH, row, lane);
            aL[mt] = load_frag_a(sAL, row, lane);
        }
#pragma unroll
        for (int nt = 0; nt < 2; ++nt) {
            int row = wn * 32 + nt * 16 + (lane & 15);
            bH[nt] = load_frag_b(sBH, row, lane);
            bL[nt] = load_frag_b(sBL, row, lane);
        }
#pragma unroll
        for (int mt = 0; mt < 2; ++mt)
#pragma unroll
            for (int nt = 0; nt < 2; ++nt) {
                acc[mt][nt] = WMMA_BF16(aH[mt], bH[nt], acc[mt][nt]);
                acc[mt][nt] = WMMA_BF16(aH[mt], bL[nt], acc[mt][nt]);
                acc[mt][nt] = WMMA_BF16(aL[mt], bH[nt], acc[mt][nt]);
            }
    }

    // ---- epilogue: bias + relu -> bf16 hi/lo activation planes
    const int m_in = ((lane >> 4) & 1) * 8;
#pragma unroll
    for (int mt = 0; mt < 2; ++mt)
#pragma unroll
        for (int nt = 0; nt < 2; ++nt) {
            int co = n0 + wn * 32 + nt * 16 + (lane & 15);
            float bias = bb[co];
#pragma unroll
            for (int r = 0; r < 8; ++r) {
                int wpix = wm * 32 + mt * 16 + m_in + r;
                float v = acc[mt][nt][r] + bias;
                v = v > 0.f ? v : 0.f;
                unsigned short hi = f2bf_bits(v);
                unsigned short lo = f2bf_bits(v - bf2f_bits(hi));
                size_t off = (((size_t)batch * NPIX) + h * FW + wpix) * CMID + co;
                baseHi[off] = hi;
                baseLo[off] = lo;
            }
        }
}

// =====================================================================
// Kernel 2: fused 1x1 heads (cls 18 + reg 36 -> N=64 padded), WMMA GEMM.
//   Both A (activation) and B (weight) tiles are TDM-fed from bf16 planes.
//   grid: (M-tiles=64, batch=8), block 128 (4 waves 2x2)
// =====================================================================
__global__ __launch_bounds__(128) void head1x1_wmma(
    const ushort_t* __restrict__ baseHi, const ushort_t* __restrict__ baseLo,
    const ushort_t* __restrict__ WoHi, const ushort_t* __restrict__ WoLo,
    const float* __restrict__ bc, const float* __restrict__ br,
    float* __restrict__ outS, float* __restrict__ outR)
{
    __shared__ __align__(16) unsigned short sAH[64 * LDSTRIDE];
    __shared__ __align__(16) unsigned short sAL[64 * LDSTRIDE];
    __shared__ __align__(16) unsigned short sBH[64 * LDSTRIDE];
    __shared__ __align__(16) unsigned short sBL[64 * LDSTRIDE];

    const int tid   = threadIdx.x;
    const int batch = blockIdx.y;
    const int m0    = blockIdx.x * 64;

    const int wave = tid >> 5, lane = tid & 31;
    const int wm = wave >> 1;      // 0..1
    const int wn = wave & 1;       // 0..1

    const unsigned sAH_off = lds_off(sAH);
    const unsigned sAL_off = lds_off(sAL);
    const unsigned sBH_off = lds_off(sBH);
    const unsigned sBL_off = lds_off(sBL);

    const v8f vzero = {0.f, 0.f, 0.f, 0.f, 0.f, 0.f, 0.f, 0.f};
    v8f acc[2][2];
#pragma unroll
    for (int mt = 0; mt < 2; ++mt)
#pragma unroll
        for (int nt = 0; nt < 2; ++nt) acc[mt][nt] = vzero;

    for (int kb = 0; kb < CMID / 32; ++kb) {
        __syncthreads();
        if (wave == 0) {
            const size_t aoff = ((size_t)batch * NPIX + m0) * CMID + kb * 32;
            tdm_load_2d_bf16(baseHi + aoff, sAH_off, 32, 64, CMID, NPIX, CMID);
            tdm_load_2d_bf16(baseLo + aoff, sAL_off, 32, 64, CMID, NPIX, CMID);
            const size_t boff = (size_t)kb * 32;
            tdm_load_2d_bf16(WoHi + boff, sBH_off, 32, 64, CMID, 64, CMID);
            tdm_load_2d_bf16(WoLo + boff, sBL_off, 32, 64, CMID, 64, CMID);
            __builtin_amdgcn_s_wait_tensorcnt(0);
        }
        __syncthreads();

        v16bf aH[2], aL[2], bH[2], bL[2];
#pragma unroll
        for (int mt = 0; mt < 2; ++mt) {
            int row = wm * 32 + mt * 16 + (lane & 15);
            aH[mt] = load_frag_a(sAH, row, lane);
            aL[mt] = load_frag_a(sAL, row, lane);
        }
#pragma unroll
        for (int nt = 0; nt < 2; ++nt) {
            int row = wn * 32 + nt * 16 + (lane & 15);
            bH[nt] = load_frag_b(sBH, row, lane);
            bL[nt] = load_frag_b(sBL, row, lane);
        }
#pragma unroll
        for (int mt = 0; mt < 2; ++mt)
#pragma unroll
            for (int nt = 0; nt < 2; ++nt) {
                acc[mt][nt] = WMMA_BF16(aH[mt], bH[nt], acc[mt][nt]);
                acc[mt][nt] = WMMA_BF16(aH[mt], bL[nt], acc[mt][nt]);
                acc[mt][nt] = WMMA_BF16(aL[mt], bH[nt], acc[mt][nt]);
            }
    }

    const int m_in = ((lane >> 4) & 1) * 8;
#pragma unroll
    for (int mt = 0; mt < 2; ++mt)
#pragma unroll
        for (int nt = 0; nt < 2; ++nt) {
            int o = wn * 32 + nt * 16 + (lane & 15);
            float bias = 0.f;
            if (o < 18)      bias = bc[o];
            else if (o < 54) bias = br[o - 18];
#pragma unroll
            for (int r = 0; r < 8; ++r) {
                int m = m0 + wm * 32 + mt * 16 + m_in + r;      // pixel index h*64+w
                float v = acc[mt][nt][r] + bias;
                if (o < 18) {
                    int a = o >> 1, c = o & 1;
                    outS[((size_t)batch * NANCH + (size_t)m * 9 + a) * 2 + c] = v;
                } else if (o < 54) {
                    int oo = o - 18;
                    int a = oo >> 2, j = oo & 3;
                    outR[((size_t)batch * NANCH + (size_t)m * 9 + a) * 4 + j] = v;
                }
            }
        }
}

// =====================================================================
// Kernel 3: softmax-fg + anchor regen + decode + clip + min-size mask
// =====================================================================
__global__ __launch_bounds__(256) void prep_proposals(
    const float* __restrict__ outS, const float* __restrict__ outR,
    const int* __restrict__ ihp, const int* __restrict__ iwp,
    float* __restrict__ boxes, float* __restrict__ scores)
{
    int gid = blockIdx.x * 256 + threadIdx.x;
    if (gid >= NIMG * NANCH) return;
    int n = gid / NANCH;
    int i = gid - n * NANCH;
    int m = i / 9;
    int a = i - m * 9;
    int hp = m >> 6, wp = m & 63;

    // ---- base anchor (numpy-equivalent, banker's rounding via rint)
    const double ratios[3] = {0.5, 1.0, 2.0};
    const double scales[3] = {8.0, 16.0, 32.0};
    int ri = a / 3, si = a - ri * 3;
    double sr  = 256.0 / ratios[ri];
    double wsd = rint(sqrt(sr));
    double hsd = rint(wsd * ratios[ri]);
    double Wd = wsd * scales[si], Hd = hsd * scales[si];
    float ax1 = (float)(7.5 - 0.5 * (Wd - 1.0)) + wp * 16.f;
    float ay1 = (float)(7.5 - 0.5 * (Hd - 1.0)) + hp * 16.f;
    float ax2 = (float)(7.5 + 0.5 * (Wd - 1.0)) + wp * 16.f;
    float ay2 = (float)(7.5 + 0.5 * (Hd - 1.0)) + hp * 16.f;

    float wa = ax2 - ax1 + 1.f, ha = ay2 - ay1 + 1.f;
    float cxa = ax1 + 0.5f * wa, cya = ay1 + 0.5f * ha;

    const float* d = outR + ((size_t)n * NANCH + i) * 4;
    float dx = d[0], dy = d[1];
    float dw = fminf(fmaxf(d[2], -4.f), 4.f);
    float dh = fminf(fmaxf(d[3], -4.f), 4.f);
    float cx = dx * wa + cxa, cy = dy * ha + cya;
    float bw = wa * expf(dw), bh = ha * expf(dh);

    float imw = (float)iwp[0], imh = (float)ihp[0];
    float b0 = fminf(fmaxf(cx - 0.5f * bw, 0.f), imw - 1.f);
    float b1 = fminf(fmaxf(cy - 0.5f * bh, 0.f), imh - 1.f);
    float b2 = fminf(fmaxf(cx + 0.5f * bw, 0.f), imw - 1.f);
    float b3 = fminf(fmaxf(cy + 0.5f * bh, 0.f), imh - 1.f);
    bool valid = ((b2 - b0 + 1.f) >= 16.f) && ((b3 - b1 + 1.f) >= 16.f);

    const float* s2 = outS + ((size_t)n * NANCH + i) * 2;
    float s0 = s2[0], s1 = s2[1];
    float mx = fmaxf(s0, s1);
    float e0 = expf(s0 - mx), e1 = expf(s1 - mx);
    float fg = e1 / (e0 + e1);

    float4 b4; b4.x = b0; b4.y = b1; b4.z = b2; b4.w = b3;
    ((float4*)boxes)[(size_t)n * NANCH + i] = b4;
    scores[(size_t)n * NANCH + i] = valid ? fg : NEG_INF_F;
}

// ---------------- float -> sortable uint (monotonic increasing) ----------------
__device__ __forceinline__ unsigned flipKey(float f) {
    unsigned u = __float_as_uint(f);
    return (u & 0x80000000u) ? ~u : (u | 0x80000000u);
}

// =====================================================================
// Kernel 4: exact top-1000 per image (radix select + bitonic sort).
// Ties broken by lowest index, matching lax.top_k.
// =====================================================================
__global__ __launch_bounds__(1024) void topk_select(
    const float* __restrict__ scores, const float* __restrict__ boxes,
    float* __restrict__ topbox, float* __restrict__ topscore)
{
    const int n = blockIdx.x;
    const int tid = threadIdx.x;
    const float* sc = scores + (size_t)n * NANCH;
    const float* bx = boxes + (size_t)n * NANCH * 4;

    __shared__ unsigned long long items[1024];
    __shared__ int scanb[1024];
    __shared__ unsigned hist[256];
    __shared__ unsigned sPrefix, sGreater, sCntG, sEBase;

    if (tid == 0) { sPrefix = 0u; sGreater = 0u; sCntG = 0u; sEBase = 0u; }
    __syncthreads();

    // ---- 4-pass radix select for the 1000th largest key
    for (int p = 3; p >= 0; --p) {
        for (int b = tid; b < 256; b += 1024) hist[b] = 0u;
        __syncthreads();
        unsigned pmask = (p == 3) ? 0u : (0xFFFFFFFFu << ((p + 1) * 8));
        unsigned pref = sPrefix;
        for (int i = tid; i < NANCH; i += 1024) {
            unsigned key = flipKey(sc[i]);
            if ((key & pmask) == (pref & pmask))
                atomicAdd(&hist[(key >> (p * 8)) & 255u], 1u);
        }
        __syncthreads();
        if (tid == 0) {
            unsigned rem = (unsigned)PRE_NMS - sGreater;
            unsigned cum = 0; int chosen = 0;
            for (int b = 255; b >= 0; --b) {
                if (cum + hist[b] >= rem) { chosen = b; break; }
                cum += hist[b];
            }
            sGreater += cum;
            sPrefix |= ((unsigned)chosen) << (p * 8);
        }
        __syncthreads();
    }
    const unsigned T = sPrefix;
    const unsigned cg = sGreater;            // count strictly greater  (<=999)
    const unsigned take = (unsigned)PRE_NMS - cg;

    // ---- compact strictly-greater (order fixed later by sort)
    for (int i = tid; i < NANCH; i += 1024) {
        unsigned key = flipKey(sc[i]);
        if (key > T) {
            unsigned p = atomicAdd(&sCntG, 1u);
            items[p] = (((unsigned long long)(~key)) << 32) | (unsigned)i;
        }
    }
    __syncthreads();

    // ---- compact ==T in ascending-index order, keep first `take`
    for (int c0 = 0; c0 < NANCH; c0 += 1024) {
        int i = c0 + tid;
        int flag = 0; unsigned key = 0;
        if (i < NANCH) { key = flipKey(sc[i]); flag = (key == T) ? 1 : 0; }
        scanb[tid] = flag;
        __syncthreads();
        for (int off = 1; off < 1024; off <<= 1) {
            int add = (tid >= off) ? scanb[tid - off] : 0;
            __syncthreads();
            scanb[tid] += add;
            __syncthreads();
        }
        if (flag) {
            unsigned pos = sEBase + (unsigned)(scanb[tid] - 1);
            if (pos < take)
                items[cg + pos] = (((unsigned long long)(~key)) << 32) | (unsigned)i;
        }
        __syncthreads();
        if (tid == 0) sEBase += (unsigned)scanb[1023];
        __syncthreads();
    }
    if (tid >= PRE_NMS) items[tid] = 0xFFFFFFFFFFFFFFFFull;
    __syncthreads();

    // ---- bitonic ascending sort of 1024 u64 -> (score desc, idx asc)
    for (unsigned ksz = 2; ksz <= 1024; ksz <<= 1) {
        for (unsigned j = ksz >> 1; j > 0; j >>= 1) {
            unsigned ixj = (unsigned)tid ^ j;
            if (ixj > (unsigned)tid) {
                unsigned long long va = items[tid], vb = items[ixj];
                bool up = (((unsigned)tid & ksz) == 0);
                if ((va > vb) == up) { items[tid] = vb; items[ixj] = va; }
            }
            __syncthreads();
        }
    }

    if (tid < PRE_NMS) {
        unsigned idx = (unsigned)(items[tid] & 0xFFFFFFFFull);
        topscore[(size_t)n * PRE_NMS + tid] = sc[idx];
        float4 b4 = ((const float4*)bx)[idx];
        ((float4*)(topbox + (size_t)n * PRE_NMS * 4))[tid] = b4;
    }
}

// =====================================================================
// Kernel 5: greedy NMS (exact reference semantics) + roi packing + count
// =====================================================================
__global__ __launch_bounds__(1024) void nms_kernel(
    const float* __restrict__ topbox, const float* __restrict__ topscore,
    float* __restrict__ rois, float* __restrict__ counts)
{
    const int n = blockIdx.x;
    const int tid = threadIdx.x;
    __shared__ float X1[1024], Y1[1024], X2[1024], Y2[1024], AR[1024];
    __shared__ int keep[1024];
    __shared__ int scanb[1024];

    if (tid < PRE_NMS) {
        float4 b = ((const float4*)(topbox + (size_t)n * PRE_NMS * 4))[tid];
        X1[tid] = b.x; Y1[tid] = b.y; X2[tid] = b.z; Y2[tid] = b.w;
        AR[tid] = (b.z - b.x + 1.f) * (b.w - b.y + 1.f);
        keep[tid] = 1;
    } else keep[tid] = 0;
    __syncthreads();

    for (int i = 0; i < PRE_NMS; ++i) {
        if (keep[i]) {
            int j = tid;
            if (j > i && j < PRE_NMS && keep[j]) {
                float ix1 = fmaxf(X1[i], X1[j]), iy1 = fmaxf(Y1[i], Y1[j]);
                float ix2 = fminf(X2[i], X2[j]), iy2 = fminf(Y2[i], Y2[j]);
                float inter = fmaxf(ix2 - ix1 + 1.f, 0.f) * fmaxf(iy2 - iy1 + 1.f, 0.f);
                float iou = inter / (AR[i] + AR[j] - inter);
                if (iou > NMS_THR) keep[j] = 0;
            }
        }
        __syncthreads();
    }

    // AND with score validity (done after NMS, like the reference)
    if (tid < PRE_NMS) {
        float s = topscore[(size_t)n * PRE_NMS + tid];
        if (!(s > NEG_INF_F * 0.5f)) keep[tid] = 0;
    }
    __syncthreads();

    scanb[tid] = keep[tid];
    __syncthreads();
    for (int off = 1; off < 1024; off <<= 1) {
        int add = (tid >= off) ? scanb[tid - off] : 0;
        __syncthreads();
        scanb[tid] += add;
        __syncthreads();
    }
    int total = scanb[1023];

    float* r = rois + (size_t)n * POST_NMS * 4;
    for (int t = tid; t < POST_NMS * 4; t += 1024) r[t] = 0.f;
    __syncthreads();
    if (tid < PRE_NMS && keep[tid]) {
        int rank = scanb[tid] - 1;
        if (rank < POST_NMS) {
            r[rank * 4 + 0] = X1[tid];
            r[rank * 4 + 1] = Y1[tid];
            r[rank * 4 + 2] = X2[tid];
            r[rank * 4 + 3] = Y2[tid];
        }
    }
    if (tid == 0) counts[n] = (float)(total < POST_NMS ? total : POST_NMS);
}

// =====================================================================
extern "C" void kernel_launch(void* const* d_in, const int* in_sizes, int n_in,
                              void* d_out, int out_size, void* d_ws, size_t ws_size,
                              hipStream_t stream)
{
    const float* x  = (const float*)d_in[0];
    const float* Wb = (const float*)d_in[1];
    const float* bb = (const float*)d_in[2];
    const float* Wc = (const float*)d_in[3];
    const float* bc = (const float*)d_in[4];
    const float* Wr = (const float*)d_in[5];
    const float* br = (const float*)d_in[6];
    const int*   ih = (const int*)d_in[7];
    const int*   iw = (const int*)d_in[8];

    // workspace carve-up (~150 MB)
    const size_t NX = (size_t)NIMG * CIN * FH * FW;     // 16,777,216
    const size_t NW = (size_t)CMID * KDIM;              // 2,359,296
    const size_t NO = (size_t)64 * CMID;                // 32,768
    const size_t NB = (size_t)NIMG * NPIX * CMID;       // 16,777,216
    ushort_t* xHi    = (ushort_t*)d_ws;
    ushort_t* xLo    = xHi + NX;
    ushort_t* WbHi   = xLo + NX;
    ushort_t* WbLo   = WbHi + NW;
    ushort_t* WoHi   = WbLo + NW;
    ushort_t* WoLo   = WoHi + NO;
    ushort_t* baseHi = WoLo + NO;
    ushort_t* baseLo = baseHi + NB;
    float* boxes    = (float*)(baseLo + NB);            // 8*36864*4
    float* scores   = boxes + (size_t)NIMG * NANCH * 4; // 8*36864
    float* topbox   = scores + (size_t)NIMG * NANCH;    // 8*1000*4
    float* topscore = topbox + (size_t)NIMG * PRE_NMS * 4;

    // output carve-up (tuple concatenated flat, all f32)
    float* outS    = (float*)d_out;                                   // 8*36864*2
    float* outR    = outS + (size_t)NIMG * NANCH * 2;                 // 8*36864*4
    float* outRois = outR + (size_t)NIMG * NANCH * 4;                 // 8*300*4
    float* outCnt  = outRois + (size_t)NIMG * POST_NMS * 4;           // 8

    // 0) one-shot bf16 hi/lo splits (bandwidth bound)
    split_pair<<<(int)((NX + 255) / 256), 256, 0, stream>>>(x, xHi, xLo, (int)NX);
    split_pair<<<(int)((NW + 255) / 256), 256, 0, stream>>>(Wb, WbHi, WbLo, (int)NW);
    build_head_w<<<(int)((NO + 255) / 256), 256, 0, stream>>>(Wc, Wr, WoHi, WoLo);

    // 1) conv + relu (WMMA + TDM weight tiles)
    dim3 g1(CMID / 128, NPIX / 64, NIMG);
    conv3x3_wmma<<<g1, 256, 0, stream>>>(xHi, xLo, WbHi, WbLo, bb, baseHi, baseLo);

    // 2) heads (WMMA, fully TDM-fed)
    dim3 g2(NPIX / 64, NIMG);
    head1x1_wmma<<<g2, 128, 0, stream>>>(baseHi, baseLo, WoHi, WoLo, bc, br, outS, outR);

    // 3..5) proposals
    prep_proposals<<<(NIMG * NANCH + 255) / 256, 256, 0, stream>>>(
        outS, outR, ih, iw, boxes, scores);
    topk_select<<<NIMG, 1024, 0, stream>>>(scores, boxes, topbox, topscore);
    nms_kernel<<<NIMG, 1024, 0, stream>>>(topbox, topscore, outRois, outCnt);
}
